// HeadNeRFModule_463856468714
// MI455X (gfx1250) — matrix operand order repaired and checked by reading, not compiled, not run on hardware
//
#include <hip/hip_runtime.h>
#include <hip/hip_bf16.h>

typedef __attribute__((ext_vector_type(16))) _Float16 v16h;
typedef __attribute__((ext_vector_type(8)))  _Float16 v8h;
typedef __attribute__((ext_vector_type(8)))  float    v8f;

// ---------------------------------------------------------------------------
// WMMA helpers (CDNA5, wave32): D = A(16x32 f16) * B(32x16 f16) + C(16x16 f32)
// ---------------------------------------------------------------------------
__device__ __forceinline__ v8f wmma_f16(v16h a, v16h b, v8f c) {
  return __builtin_amdgcn_wmma_f32_16x16x32_f16(
      /*neg_a=*/false, a, /*neg_b=*/false, b,
      /*c_mod=*/(short)0, c, /*reuse_a=*/false, /*reuse_b=*/false);
}

__device__ __forceinline__ v16h frag_cat(v8h lo, v8h hi) {
  v16h r;
#pragma unroll
  for (int i = 0; i < 8; ++i) { r[i] = lo[i]; r[8 + i] = hi[i]; }
  return r;
}

// A-matrix 16x32 f16 fragment from LDS row-major activations [16][stride].
// Lane L: m = L&15, h = L>>4 holds K = kbase+8h..+7 and kbase+16+8h..+7
// (two contiguous aligned 16B runs -> ds_load_b128 x2).
__device__ __forceinline__ v16h load_a_frag(const _Float16* act, int stride,
                                            int kbase, int lane) {
  int m = lane & 15, h = lane >> 4;
  const _Float16* p = act + m * stride + kbase + 8 * h;
  v8h lo = *(const v8h*)p;
  v8h hi = *(const v8h*)(p + 16);
  return frag_cat(lo, hi);
}

// B-matrix 32x16 f16 fragment from LDS weights stored TRANSPOSED [Nout][Kp].
// Lane L: n = L&15, g = L>>4 holds K = kbase+8g..+7 and kbase+16+8g..+7 of column ncol0+n
// (two contiguous aligned 16B runs -> ds_load_b128 x2).
__device__ __forceinline__ v16h load_b_frag(const _Float16* Wt, int Kp,
                                            int ncol0, int kbase, int lane) {
  int n = lane & 15, g = lane >> 4;
  const _Float16* p = Wt + (size_t)(ncol0 + n) * Kp + kbase + 8 * g;
  v8h lo = *(const v8h*)p;
  v8h hi = *(const v8h*)(p + 16);
  return frag_cat(lo, hi);
}

// ---------------------------------------------------------------------------
// Trilinear sampling of a cubic volume (C channels, base dim D0, subsample s),
// border-zero semantics matching grid_sample3d. Volumes are L2-resident.
// ---------------------------------------------------------------------------
template <int C>
__device__ __forceinline__ void sample3(const float* __restrict__ vol, int D0,
                                        int s, int Ds, float ux, float uy,
                                        float uz, float* out) {
  const int A = D0 * D0;
  float fx = ((ux + 1.f) * (float)Ds - 1.f) * 0.5f;
  float fy = ((uy + 1.f) * (float)Ds - 1.f) * 0.5f;
  float fz = ((uz + 1.f) * (float)Ds - 1.f) * 0.5f;
  float xf = floorf(fx), yf = floorf(fy), zf = floorf(fz);
  float wx = fx - xf, wy = fy - yf, wz = fz - zf;
  int x0 = (int)xf, y0 = (int)yf, z0 = (int)zf;
#pragma unroll
  for (int c = 0; c < C; ++c) out[c] = 0.f;
#pragma unroll
  for (int dz = 0; dz < 2; ++dz) {
    int zi = z0 + dz;
    if ((unsigned)zi >= (unsigned)Ds) continue;
    float wzz = dz ? wz : 1.f - wz;
#pragma unroll
    for (int dy = 0; dy < 2; ++dy) {
      int yi = y0 + dy;
      if ((unsigned)yi >= (unsigned)Ds) continue;
      float wyy = dy ? wy : 1.f - wy;
#pragma unroll
      for (int dx = 0; dx < 2; ++dx) {
        int xi = x0 + dx;
        if ((unsigned)xi >= (unsigned)Ds) continue;
        float w = wzz * wyy * (dx ? wx : 1.f - wx);
        int base = (zi * s) * A + (yi * s) * D0 + (xi * s);
#pragma unroll
        for (int c = 0; c < C; ++c) out[c] += vol[c * D0 * A + base] * w;
      }
    }
  }
}

// Positional embedding fill: [x(C), sin(x)(C), cos(x)(C), sin(2x)... k<4]
// via double-angle recurrence (1 sin + 1 cos per channel instead of 8).
__device__ __forceinline__ void fill_fe24(const float* baseF, int bstride,
                                          _Float16* actA, int astride,
                                          int lane) {
  for (int t = lane; t < 16 * 24; t += 32) {
    int m = t / 24, ch = t - m * 24;
    float v = baseF[m * bstride + ch];
    _Float16* row = actA + m * astride;
    row[ch] = (_Float16)v;
    float s = __sinf(v), c = __cosf(v);
#pragma unroll
    for (int k = 0; k < 4; ++k) {
      row[24 + 48 * k + ch]      = (_Float16)s;
      row[24 + 48 * k + 24 + ch] = (_Float16)c;
      float s2 = 2.f * s * c, c2 = c * c - s * s;
      s = s2; c = c2;
    }
  }
}

// ---------------------------------------------------------------------------
// Prep kernels
// ---------------------------------------------------------------------------
__global__ void k_build_deform(const float* __restrict__ expg,
                               const float* __restrict__ bs,
                               const float* __restrict__ dmean,
                               float* __restrict__ dvol, int B) {
  const int per = 4 * 32 * 32 * 32;
  int i = blockIdx.x * blockDim.x + threadIdx.x;
  if (i >= B * per) return;
  int b = i / per, v = i - b * per;
  float acc = dmean[v];
#pragma unroll 4
  for (int e = 0; e < 32; ++e) acc += expg[b * 32 + e] * bs[(size_t)e * per + v];
  dvol[i] = acc;
}

// fp32 [K][Nout] -> f16 transposed + padded [Np][Kp]
__global__ void k_pack_wt(const float* __restrict__ src, _Float16* __restrict__ dst,
                          int K, int Nout, int Kp, int NpKp) {
  int i = blockIdx.x * blockDim.x + threadIdx.x;
  if (i >= NpKp) return;
  int n = i / Kp, k = i - n * Kp;
  float v = (k < K && n < Nout) ? src[(size_t)k * Nout + n] : 0.f;
  dst[i] = (_Float16)v;
}

// cW1 (275x128) packed with ROW PERMUTATION into [128][288] so that color
// shares the density activation layout: [fe(216), exp(32), ve(27), pad(13)].
//   packed row k<216  <- cW1 row k        (fe)
//   packed row 216..247 <- cW1 row k+27   (exp rows 243..274)
//   packed row 248..274 <- cW1 row k-32   (ve rows 216..242)
__global__ void k_pack_cw1(const float* __restrict__ src, _Float16* __restrict__ dst) {
  int i = blockIdx.x * blockDim.x + threadIdx.x;
  if (i >= 128 * 288) return;
  int n = i / 288, k = i - n * 288;
  int ks;
  if (k < 216) ks = k;
  else if (k < 248) ks = k + 27;
  else if (k < 275) ks = k - 32;
  else ks = -1;
  float v = (ks >= 0) ? src[(size_t)ks * 128 + n] : 0.f;
  dst[i] = (_Float16)v;
}

// ---------------------------------------------------------------------------
// Kernel 1: deform sampling + embed(12->108) + MLP 108->64(relu)->3
//           outputs offset and deformed points. 8 waves/block, no inner
//           barriers (all staging is wave-private; LDS is in-order per wave).
// ---------------------------------------------------------------------------
__global__ __launch_bounds__(256)
void k_deform_mlp(const float* __restrict__ qpts, const float* __restrict__ dvol_all,
                  const _Float16* __restrict__ gW1, const _Float16* __restrict__ gW2,
                  const float* __restrict__ db1, const float* __restrict__ db2,
                  float* __restrict__ out_offset, float* __restrict__ dpts,
                  int Bn, int Nn) {
  extern __shared__ char smem[];
  _Float16* sW1 = (_Float16*)smem;        // [64][128]
  _Float16* sW2 = sW1 + 64 * 128;         // [16][64]
  float* fbase = (float*)(sW2 + 16 * 64);
  int wave = threadIdx.x >> 5, lane = threadIdx.x & 31;
  float* baseF = fbase + wave * 16 * 12;
  _Float16* actbase = (_Float16*)(fbase + 8 * 16 * 12);
  _Float16* actA = actbase + wave * 16 * 128;
  _Float16* actH = actbase + 8 * 16 * 128 + wave * 16 * 64;

  __builtin_prefetch(gW1, 0, 1);
  for (int i = threadIdx.x; i < 64 * 128; i += blockDim.x) sW1[i] = gW1[i];
  for (int i = threadIdx.x; i < 16 * 64; i += blockDim.x) sW2[i] = gW2[i];
  __syncthreads();

  // loop-invariant biases
  int g = lane >> 4, nlo = lane & 15;
  float bia1[4];
#pragma unroll
  for (int nt = 0; nt < 4; ++nt) bia1[nt] = db1[nt * 16 + nlo];
  float db2v = (nlo < 3) ? db2[nlo] : 0.f;

  int ntiles = (Bn * Nn) >> 4;
  int gw = blockIdx.x * (blockDim.x >> 5) + wave;
  int gws = gridDim.x * (blockDim.x >> 5);

  for (int t = gw; t < ntiles; t += gws) {
    int r0 = t << 4;
    int b = r0 / Nn, n0 = r0 - b * Nn;
    const float* dv = dvol_all + (size_t)b * 4 * 32768;

    // (point, level) sampling tasks spread over all 32 lanes
    for (int t2 = lane; t2 < 48; t2 += 32) {
      int m = t2 / 3, lvl = t2 - 3 * m;
      int n = n0 + m;
      float px = qpts[((size_t)b * 3 + 0) * Nn + n];
      float py = qpts[((size_t)b * 3 + 1) * Nn + n];
      float pz = qpts[((size_t)b * 3 + 2) * Nn + n];
      float f[4];
      sample3<4>(dv, 32, 1 << lvl, 32 >> lvl, 2.f * px - 1.f, 2.f * py - 1.f,
                 2.f * pz - 1.f, f);
#pragma unroll
      for (int c = 0; c < 4; ++c) baseF[m * 12 + lvl * 4 + c] = f[c];
    }

    // embed 12 -> 108 (pad to 128)
    for (int t2 = lane; t2 < 16 * 12; t2 += 32) {
      int m = t2 / 12, ch = t2 - m * 12;
      float v = baseF[m * 12 + ch];
      _Float16* row = actA + m * 128;
      row[ch] = (_Float16)v;
      float s = __sinf(v), c = __cosf(v);
#pragma unroll
      for (int k = 0; k < 4; ++k) {
        row[12 + 24 * k + ch]      = (_Float16)s;
        row[12 + 24 * k + 12 + ch] = (_Float16)c;
        float s2 = 2.f * s * c, c2 = c * c - s * s;
        s = s2; c = c2;
      }
    }
    for (int t2 = lane; t2 < 16 * 20; t2 += 32) {
      int m = t2 / 20;
      actA[m * 128 + 108 + (t2 - m * 20)] = (_Float16)0.f;
    }

    // layer 1: 16x128 @ 128x64
    {
      v8f acc[4];
#pragma unroll
      for (int nt = 0; nt < 4; ++nt)
#pragma unroll
        for (int r = 0; r < 8; ++r) acc[nt][r] = 0.f;
      for (int ks = 0; ks < 4; ++ks) {
        v16h a = load_a_frag(actA, 128, ks * 32, lane);
#pragma unroll
        for (int nt = 0; nt < 4; ++nt) {
          v16h bf = load_b_frag(sW1, 128, nt * 16, ks * 32, lane);
          acc[nt] = wmma_f16(a, bf, acc[nt]);
        }
      }
#pragma unroll
      for (int nt = 0; nt < 4; ++nt)
#pragma unroll
        for (int r = 0; r < 8; ++r) {
          float v = fmaxf(acc[nt][r] + bia1[nt], 0.f);
          actH[(r + 8 * g) * 64 + nt * 16 + nlo] = (_Float16)v;
        }
    }

    // layer 2: 16x64 @ 64x16 (3 valid cols)
    {
      v8f acc;
#pragma unroll
      for (int r = 0; r < 8; ++r) acc[r] = 0.f;
      for (int ks = 0; ks < 2; ++ks) {
        v16h a = load_a_frag(actH, 64, ks * 32, lane);
        v16h bf = load_b_frag(sW2, 64, 0, ks * 32, lane);
        acc = wmma_f16(a, bf, acc);
      }
      if (nlo < 3) {
#pragma unroll
        for (int r = 0; r < 8; ++r) {
          int m = r + 8 * g;
          size_t oi = ((size_t)b * 3 + nlo) * Nn + (n0 + m);
          float off = acc[r] + db2v;
          out_offset[oi] = off;
          dpts[oi] = off * 0.1f + qpts[oi];
        }
      }
    }
  }
}

// ---------------------------------------------------------------------------
// Kernel 2 (merged): feature sampling + embed done ONCE, shared activation
// matrix [fe(216), exp(32), ve(27), pad] feeds BOTH MLPs:
//   density: k-steps 0..7  (ve cols hit zero weight rows)  256->128(relu)->1
//   color:   k-steps 0..8                                  288->128(relu)->3
// 8 waves/block; no barriers in the tile loop (wave-private staging).
// ---------------------------------------------------------------------------
__global__ __launch_bounds__(256)
void k_density_color(const float* __restrict__ dpts, const float* __restrict__ vdirs,
                     const float* __restrict__ fvol, const float* __restrict__ expg,
                     const _Float16* __restrict__ gsW1, const _Float16* __restrict__ gsW2,
                     const _Float16* __restrict__ gcW1, const _Float16* __restrict__ gcW2,
                     const float* __restrict__ sb1, const float* __restrict__ sb2,
                     const float* __restrict__ cb1, const float* __restrict__ cb2,
                     float* __restrict__ out_d, float* __restrict__ out_c,
                     int Bn, int Nn) {
  extern __shared__ char smem[];
  _Float16* sW1s = (_Float16*)smem;          // [128][256]
  _Float16* sW2s = sW1s + 128 * 256;         // [16][128]
  _Float16* cW1s = sW2s + 16 * 128;          // [128][288]
  _Float16* cW2s = cW1s + 128 * 288;         // [16][128]
  float* fbase = (float*)(cW2s + 16 * 128);
  int wave = threadIdx.x >> 5, lane = threadIdx.x & 31;
  float* baseF = fbase + wave * 16 * 27;     // 24 feats + 3 viewdirs
  _Float16* actbase = (_Float16*)(fbase + 8 * 16 * 27);
  _Float16* actA = actbase + wave * 16 * 288;
  _Float16* actH = actbase + 8 * 16 * 288 + wave * 16 * 128;

  __builtin_prefetch(gsW1, 0, 1);
  __builtin_prefetch(gcW1, 0, 1);
  for (int i = threadIdx.x; i < 128 * 256; i += blockDim.x) sW1s[i] = gsW1[i];
  for (int i = threadIdx.x; i < 16 * 128; i += blockDim.x) sW2s[i] = gsW2[i];
  for (int i = threadIdx.x; i < 128 * 288; i += blockDim.x) cW1s[i] = gcW1[i];
  for (int i = threadIdx.x; i < 16 * 128; i += blockDim.x) cW2s[i] = gcW2[i];
  __syncthreads();

  // loop-invariant biases
  int g = lane >> 4, nlo = lane & 15;
  float biaS[8], biaC[8];
#pragma unroll
  for (int nt = 0; nt < 8; ++nt) {
    biaS[nt] = sb1[nt * 16 + nlo];
    biaC[nt] = cb1[nt * 16 + nlo];
  }
  float sb2v = sb2[0];
  float cb2v = (nlo < 3) ? cb2[nlo] : 0.f;

  int ntiles = (Bn * Nn) >> 4;
  int gw = blockIdx.x * (blockDim.x >> 5) + wave;
  int gws = gridDim.x * (blockDim.x >> 5);

  for (int t = gw; t < ntiles; t += gws) {
    int r0 = t << 4;
    int b = r0 / Nn, n0 = r0 - b * Nn;

    // (point, level) sampling tasks spread over all 32 lanes
    for (int t2 = lane; t2 < 48; t2 += 32) {
      int m = t2 / 3, lvl = t2 - 3 * m;
      int n = n0 + m;
      float px = dpts[((size_t)b * 3 + 0) * Nn + n];
      float py = dpts[((size_t)b * 3 + 1) * Nn + n];
      float pz = dpts[((size_t)b * 3 + 2) * Nn + n];
      float f[8];
      sample3<8>(fvol, 64, 1 << lvl, 64 >> lvl, 2.f * px - 1.f, 2.f * py - 1.f,
                 2.f * pz - 1.f, f);
#pragma unroll
      for (int c = 0; c < 8; ++c) baseF[m * 27 + lvl * 8 + c] = f[c];
    }
    if (lane < 16) {
      int n = n0 + lane;
#pragma unroll
      for (int c = 0; c < 3; ++c)
        baseF[lane * 27 + 24 + c] = vdirs[((size_t)b * 3 + c) * Nn + n];
    }

    // fe embed: cols [0, 216)
    fill_fe24(baseF, 27, actA, 288, lane);
    // exp: cols [216, 248) -- column index == lane, value invariant over rows
    {
      _Float16 hv = (_Float16)(expg[b * 32 + lane] * (1.f / 3.f));
#pragma unroll
      for (int j = 0; j < 16; ++j) actA[j * 288 + 216 + lane] = hv;
    }
    // viewdir embed: cols [248, 275)
    for (int t2 = lane; t2 < 48; t2 += 32) {
      int m = t2 / 3, ch = t2 - 3 * m;
      float v = baseF[m * 27 + 24 + ch];
      _Float16* row = actA + m * 288;
      row[248 + ch] = (_Float16)v;
      float s = __sinf(v), c = __cosf(v);
#pragma unroll
      for (int k = 0; k < 4; ++k) {
        row[251 + 6 * k + ch] = (_Float16)s;
        row[254 + 6 * k + ch] = (_Float16)c;
        float s2 = 2.f * s * c, c2 = c * c - s * s;
        s = s2; c = c2;
      }
    }
    // pad: cols [275, 288)
    for (int t2 = lane; t2 < 16 * 13; t2 += 32) {
      int m = t2 / 13;
      actA[m * 288 + 275 + (t2 - 13 * m)] = (_Float16)0.f;
    }

    // ---- density layer 1: 16x256 @ 256x128 (k-steps 0..7 of shared actA)
    {
      v8f acc[8];
#pragma unroll
      for (int nt = 0; nt < 8; ++nt)
#pragma unroll
        for (int r = 0; r < 8; ++r) acc[nt][r] = 0.f;
      for (int ks = 0; ks < 8; ++ks) {
        v16h a = load_a_frag(actA, 288, ks * 32, lane);
#pragma unroll
        for (int nt = 0; nt < 8; ++nt) {
          v16h bf = load_b_frag(sW1s, 256, nt * 16, ks * 32, lane);
          acc[nt] = wmma_f16(a, bf, acc[nt]);
        }
      }
#pragma unroll
      for (int nt = 0; nt < 8; ++nt)
#pragma unroll
        for (int r = 0; r < 8; ++r) {
          float v = fmaxf(acc[nt][r] + biaS[nt], 0.f);
          actH[(r + 8 * g) * 128 + nt * 16 + nlo] = (_Float16)v;
        }
    }
    // ---- density layer 2: 16x128 @ 128x16 (1 valid col)
    {
      v8f acc;
#pragma unroll
      for (int r = 0; r < 8; ++r) acc[r] = 0.f;
      for (int ks = 0; ks < 4; ++ks) {
        v16h a = load_a_frag(actH, 128, ks * 32, lane);
        v16h bf = load_b_frag(sW2s, 128, 0, ks * 32, lane);
        acc = wmma_f16(a, bf, acc);
      }
      if (nlo == 0) {
#pragma unroll
        for (int r = 0; r < 8; ++r)
          out_d[(size_t)b * Nn + n0 + r + 8 * g] = acc[r] + sb2v;
      }
    }

    // ---- color layer 1: 16x288 @ 288x128 (actA still intact, actH reused)
    {
      v8f acc[8];
#pragma unroll
      for (int nt = 0; nt < 8; ++nt)
#pragma unroll
        for (int r = 0; r < 8; ++r) acc[nt][r] = 0.f;
      for (int ks = 0; ks < 9; ++ks) {
        v16h a = load_a_frag(actA, 288, ks * 32, lane);
#pragma unroll
        for (int nt = 0; nt < 8; ++nt) {
          v16h bf = load_b_frag(cW1s, 288, nt * 16, ks * 32, lane);
          acc[nt] = wmma_f16(a, bf, acc[nt]);
        }
      }
#pragma unroll
      for (int nt = 0; nt < 8; ++nt)
#pragma unroll
        for (int r = 0; r < 8; ++r) {
          float v = fmaxf(acc[nt][r] + biaC[nt], 0.f);
          actH[(r + 8 * g) * 128 + nt * 16 + nlo] = (_Float16)v;
        }
    }
    // ---- color layer 2: 16x128 @ 128x16 (3 valid cols), sigmoid
    {
      v8f acc;
#pragma unroll
      for (int r = 0; r < 8; ++r) acc[r] = 0.f;
      for (int ks = 0; ks < 4; ++ks) {
        v16h a = load_a_frag(actH, 128, ks * 32, lane);
        v16h bf = load_b_frag(cW2s, 128, 0, ks * 32, lane);
        acc = wmma_f16(a, bf, acc);
      }
      if (nlo < 3) {
#pragma unroll
        for (int r = 0; r < 8; ++r) {
          float v = acc[r] + cb2v;
          v = 1.f / (1.f + __expf(-v));
          out_c[((size_t)b * 3 + nlo) * Nn + n0 + r + 8 * g] = v;
        }
      }
    }
  }
}

// ---------------------------------------------------------------------------
// Host launcher
// ---------------------------------------------------------------------------
extern "C" void kernel_launch(void* const* d_in, const int* in_sizes, int n_in,
                              void* d_out, int out_size, void* d_ws, size_t ws_size,
                              hipStream_t stream) {
  const float* qpts  = (const float*)d_in[0];
  const float* vdirs = (const float*)d_in[1];
  const float* expg  = (const float*)d_in[2];
  const float* bs    = (const float*)d_in[3];
  const float* dmean = (const float*)d_in[4];
  const float* fvol  = (const float*)d_in[5];
  const float* dW1 = (const float*)d_in[6];
  const float* db1 = (const float*)d_in[7];
  const float* dW2 = (const float*)d_in[8];
  const float* db2 = (const float*)d_in[9];
  const float* sW1 = (const float*)d_in[10];
  const float* sb1 = (const float*)d_in[11];
  const float* sW2 = (const float*)d_in[12];
  const float* sb2 = (const float*)d_in[13];
  const float* cW1 = (const float*)d_in[14];
  const float* cb1 = (const float*)d_in[15];
  const float* cW2 = (const float*)d_in[16];
  const float* cb2 = (const float*)d_in[17];

  int B = in_sizes[2] / 32;          // exp is (B, 32)
  int N = in_sizes[0] / (3 * B);     // query_pts is (B, 3, N)

  // Workspace layout
  float* dvol = (float*)d_ws;                                  // B*4*32^3 f32
  float* dpts = dvol + (size_t)B * 4 * 32 * 32 * 32;           // B*3*N f32
  _Float16* w = (_Float16*)(dpts + (size_t)B * 3 * N);
  _Float16* dW1t = w;  w += 64 * 128;
  _Float16* dW2t = w;  w += 16 * 64;
  _Float16* sW1t = w;  w += 128 * 256;
  _Float16* sW2t = w;  w += 16 * 128;
  _Float16* cW1t = w;  w += 128 * 288;
  _Float16* cW2t = w;  w += 16 * 128;

  float* out_offset  = (float*)d_out;
  float* out_density = out_offset + (size_t)B * 3 * N;
  float* out_color   = out_density + (size_t)B * N;

  int nv = B * 4 * 32 * 32 * 32;
  k_build_deform<<<(nv + 255) / 256, 256, 0, stream>>>(expg, bs, dmean, dvol, B);
  k_pack_wt<<<(64 * 128 + 255) / 256, 256, 0, stream>>>(dW1, dW1t, 108, 64, 128, 64 * 128);
  k_pack_wt<<<(16 * 64 + 255) / 256, 256, 0, stream>>>(dW2, dW2t, 64, 3, 64, 16 * 64);
  k_pack_wt<<<(128 * 256 + 255) / 256, 256, 0, stream>>>(sW1, sW1t, 248, 128, 256, 128 * 256);
  k_pack_wt<<<(16 * 128 + 255) / 256, 256, 0, stream>>>(sW2, sW2t, 128, 1, 128, 16 * 128);
  k_pack_cw1<<<(128 * 288 + 255) / 256, 256, 0, stream>>>(cW1, cW1t);
  k_pack_wt<<<(16 * 128 + 255) / 256, 256, 0, stream>>>(cW2, cW2t, 128, 3, 128, 16 * 128);

  // 8 waves/block
  const size_t smem1 = (64 * 128 + 16 * 64) * 2 + 8 * 16 * 12 * 4 +
                       (8 * 16 * 128 + 8 * 16 * 64) * 2;                 // 73728
  const size_t smem2 = (128 * 256 + 16 * 128 + 128 * 288 + 16 * 128) * 2 +
                       8 * 16 * 27 * 4 +
                       (8 * 16 * 288 + 8 * 16 * 128) * 2;                // 267776

  (void)hipFuncSetAttribute((const void*)k_deform_mlp,
                            hipFuncAttributeMaxDynamicSharedMemorySize, (int)smem1);
  (void)hipFuncSetAttribute((const void*)k_density_color,
                            hipFuncAttributeMaxDynamicSharedMemorySize, (int)smem2);

  const int blocks = 512;  // grid-stride over 16-point tiles
  k_deform_mlp<<<blocks, 256, smem1, stream>>>(qpts, dvol, dW1t, dW2t, db1, db2,
                                               out_offset, dpts, B, N);
  k_density_color<<<blocks, 256, smem2, stream>>>(dpts, vdirs, fvol, expg,
                                                  sW1t, sW2t, cW1t, cW2t,
                                                  sb1, sb2, cb1, cb2,
                                                  out_density, out_color, B, N);
}
